// TokenStage_27462020891241
// MI455X (gfx1250) — compile-verified, matrix operand order checked by reading, not run on hardware
//
#include <hip/hip_runtime.h>
#include <hip/hip_bf16.h>
#include <math.h>

// ---------------- model dims ----------------
#define B_    2
#define CIN_  3
#define IMG_  128
#define P_    4
#define GS_   32
#define L_    1024
#define D_    192
#define WS_   8
#define SHIFT_ 4
#define NH_   6
#define HD_   32
#define DI_   384
#define DS_   16
#define R_    12
#define K_    4
#define NW_   16
#define W44_  (R_ + 2*DS_)   /* 44 */

typedef __attribute__((ext_vector_type(16))) _Float16 v16h;
typedef __attribute__((ext_vector_type(8)))  _Float16 v8h;
typedef __attribute__((ext_vector_type(8)))  float    v8f;

// ---------------- device helpers ----------------
__device__ __forceinline__ float sigmoidf_(float x) { return 1.f / (1.f + __expf(-x)); }
__device__ __forceinline__ float siluf_(float x)    { return x * sigmoidf_(x); }
__device__ __forceinline__ float geluf_(float x) {
  float x3 = x * x * x;
  return 0.5f * x * (1.f + tanhf(0.7978845608028654f * (x + 0.044715f * x3)));
}
__device__ __forceinline__ float act_apply(float v, int act) {
  if (act == 1) return geluf_(v);
  if (act == 2) return siluf_(v);
  return v;
}

// Load one 16-bit 16x32 WMMA fragment slice for this lane (f16 source data).
// p = row_base + 8*half + kk. Fragment element e (ISA 7.12.2):
//   e=0..7  -> k = e      (relative to p)
//   e=8..15 -> k = 16+e-8
// Two b128 loads of packed halves; pure register placement, no conversion.
__device__ __forceinline__ v16h load_frag16h(const _Float16* __restrict__ p) {
  v8h lo = *(const v8h*)(p);
  v8h hi = *(const v8h*)(p + 16);
  v16h f;
#pragma unroll
  for (int i = 0; i < 8; ++i) { f[i] = lo[i]; f[8 + i] = hi[i]; }
  return f;
}

// ---------------- WMMA GEMM (f16 inputs, f32 accumulate) ----------------
// C[m,n] = act( sum_k A[m,k] * W[n,k] + bias[n] ); A,W are f16, K-contiguous.
// Requires M % 32 == 0, Kd % 32 == 0; N tail: loads clamped, stores guarded.
// Output: f32 to C (strided scM/scN), or f16 to Ch if Ch != nullptr.
// One wave computes a 32x32 tile (2x2 WMMA blocking). grid=(M/32, ceil(N/32), batch).
__global__ void gemm_wmma_kernel(const _Float16* __restrict__ A,
                                 const _Float16* __restrict__ W,
                                 const float* __restrict__ bias,
                                 float* __restrict__ C,
                                 _Float16* __restrict__ Ch,
                                 int M, int N, int Kd,
                                 int lda,                     // A row stride (elems)
                                 int scM, int scN,            // C element strides
                                 long aBatchStride, long cBatchStride,
                                 int wBatchMod, long wBatchStride,
                                 int act) {
  const int lane = threadIdx.x & 31;
  const int tM = blockIdx.x * 32;
  const int tN = blockIdx.y * 32;
  const int z  = blockIdx.z;

  const _Float16* Ab = A + (long)z * aBatchStride;
  const _Float16* Wb = W + (long)(z % wBatchMod) * wBatchStride;

  const int half = lane >> 4;
  const int l15  = lane & 15;
  const int n0 = tN + l15;
  const int n1 = tN + 16 + l15;
  const int n0c = (n0 < N) ? n0 : (N - 1);   // clamp: safe loads, discarded on store
  const int n1c = (n1 < N) ? n1 : (N - 1);

  const _Float16* a0p = Ab + (long)(tM + l15) * lda + 8 * half;
  const _Float16* a1p = Ab + (long)(tM + 16 + l15) * lda + 8 * half;
  const _Float16* b0p = Wb + (long)n0c * Kd + 8 * half;
  const _Float16* b1p = Wb + (long)n1c * Kd + 8 * half;

  v8f acc00 = {}, acc01 = {}, acc10 = {}, acc11 = {};

  for (int kk = 0; kk < Kd; kk += 32) {
    v16h a0 = load_frag16h(a0p + kk);
    v16h a1 = load_frag16h(a1p + kk);
    v16h b0 = load_frag16h(b0p + kk);
    v16h b1 = load_frag16h(b1p + kk);
    acc00 = __builtin_amdgcn_wmma_f32_16x16x32_f16(false, a0, false, b0, (short)0, acc00, false, false);
    acc01 = __builtin_amdgcn_wmma_f32_16x16x32_f16(false, a0, false, b1, (short)0, acc01, false, false);
    acc10 = __builtin_amdgcn_wmma_f32_16x16x32_f16(false, a1, false, b0, (short)0, acc10, false, false);
    acc11 = __builtin_amdgcn_wmma_f32_16x16x32_f16(false, a1, false, b1, (short)0, acc11, false, false);
  }

  const float bn0 = (bias && n0 < N) ? bias[n0] : 0.f;
  const float bn1 = (bias && n1 < N) ? bias[n1] : 0.f;

  if (Ch) {
    _Float16* Chb = Ch + (long)z * cBatchStride;
#pragma unroll
    for (int r = 0; r < 8; ++r) {
      int mA = tM + r + 8 * half;
      int mB = mA + 16;
      if (n0 < N) {
        Chb[(long)mA * scM + (long)n0 * scN] = (_Float16)act_apply(acc00[r] + bn0, act);
        Chb[(long)mB * scM + (long)n0 * scN] = (_Float16)act_apply(acc10[r] + bn0, act);
      }
      if (n1 < N) {
        Chb[(long)mA * scM + (long)n1 * scN] = (_Float16)act_apply(acc01[r] + bn1, act);
        Chb[(long)mB * scM + (long)n1 * scN] = (_Float16)act_apply(acc11[r] + bn1, act);
      }
    }
  } else {
    float* Cb = C + (long)z * cBatchStride;
#pragma unroll
    for (int r = 0; r < 8; ++r) {
      int mA = tM + r + 8 * half;
      int mB = mA + 16;
      if (n0 < N) {
        Cb[(long)mA * scM + (long)n0 * scN] = act_apply(acc00[r] + bn0, act);
        Cb[(long)mB * scM + (long)n0 * scN] = act_apply(acc10[r] + bn0, act);
      }
      if (n1 < N) {
        Cb[(long)mA * scM + (long)n1 * scN] = act_apply(acc01[r] + bn1, act);
        Cb[(long)mB * scM + (long)n1 * scN] = act_apply(acc11[r] + bn1, act);
      }
    }
  }
}

// ---------------- f32 -> f16 conversion (weights, once per call) ----------------
__global__ void f2h_kernel(const float* __restrict__ a, _Float16* __restrict__ o, int n) {
  int idx = blockIdx.x * blockDim.x + threadIdx.x;
  if (idx < n) o[idx] = (_Float16)a[idx];
}

// ---------------- positional embedding ----------------
__global__ void pos_kernel(float* __restrict__ pos) {
  int idx = blockIdx.x * blockDim.x + threadIdx.x;
  if (idx >= L_ * D_) return;
  int d = idx % D_;
  int t = idx / D_;
  int i = t >> 5, j = t & 31;
  int c, p;
  if (d < 96) { c = d;      p = i; }
  else        { c = d - 96; p = j; }
  float div = __expf((float)(2 * (c % 48)) * (-logf(10000.f) / 96.f));
  float s = (float)p * div;
  pos[idx] = (c < 48) ? sinf(s) : cosf(s);
}

// ---------------- patch embed conv (4x4 stride 4) -> (B,L,D) ----------------
__global__ void patch_conv_kernel(const float* __restrict__ x,
                                  const float* __restrict__ w,
                                  const float* __restrict__ bias,
                                  float* __restrict__ y) {
  int idx = blockIdx.x * blockDim.x + threadIdx.x;
  if (idx >= B_ * L_ * D_) return;
  int d = idx % D_;
  int t = (idx / D_) % L_;
  int b = idx / (D_ * L_);
  int i = t >> 5, j = t & 31;
  float s = bias[d];
  for (int c = 0; c < CIN_; ++c)
    for (int py = 0; py < P_; ++py)
      for (int px = 0; px < P_; ++px)
        s += x[((long)(b * CIN_ + c) * IMG_ + (i * P_ + py)) * IMG_ + (j * P_ + px)] *
             w[((d * CIN_ + c) * P_ + py) * P_ + px];
  y[idx] = s;
}

// ---------------- LayerNorm (one wave32 per row) ----------------
// Optional f32 output yf, optional f16 output yh (for GEMM consumers), optional +alpha*pos.
__global__ void ln_kernel(const float* __restrict__ x, const float* __restrict__ g,
                          const float* __restrict__ bta,
                          float* __restrict__ yf, _Float16* __restrict__ yh,
                          int rows, int C,
                          const float* __restrict__ pos,
                          const float* __restrict__ alpha_p, int posPeriod) {
  int wave = (blockIdx.x * blockDim.x + threadIdx.x) >> 5;
  int lane = threadIdx.x & 31;
  if (wave >= rows) return;
  const float* xr = x + (long)wave * C;
  float s = 0.f, ss = 0.f;
  for (int c = lane; c < C; c += 32) { float v = xr[c]; s += v; ss += v * v; }
  for (int off = 16; off > 0; off >>= 1) {
    s  += __shfl_xor(s,  off, 32);
    ss += __shfl_xor(ss, off, 32);
  }
  float m   = s / (float)C;
  float var = ss / (float)C - m * m;
  float inv = rsqrtf(var + 1e-5f);
  float alpha = alpha_p ? alpha_p[0] : 0.f;
  for (int c = lane; c < C; c += 32) {
    float v = (xr[c] - m) * inv * g[c] + bta[c];
    if (pos) v += alpha * pos[(long)(wave % posPeriod) * C + c];
    if (yf) yf[(long)wave * C + c] = v;
    if (yh) yh[(long)wave * C + c] = (_Float16)v;
  }
}

// ---------------- depthwise 3x3 conv + SiLU: xz(B,L,2DI) -> xc(B,DI,L) ----------------
__global__ void dwconv_silu_kernel(const float* __restrict__ xz,
                                   const float* __restrict__ cw,
                                   const float* __restrict__ cb,
                                   float* __restrict__ xc) {
  int idx = blockIdx.x * blockDim.x + threadIdx.x;
  if (idx >= B_ * DI_ * L_) return;
  int l = idx % L_;
  int d = (idx / L_) % DI_;
  int b = idx / (L_ * DI_);
  int i = l >> 5, j = l & 31;
  float s = cb[d];
  for (int di = -1; di <= 1; ++di)
    for (int dj = -1; dj <= 1; ++dj) {
      int ii = i + di, jj = j + dj;
      if (ii >= 0 && ii < GS_ && jj >= 0 && jj < GS_)
        s += xz[((long)(b * L_ + ii * GS_ + jj)) * (2 * DI_) + d] *
             cw[d * 9 + (di + 1) * 3 + (dj + 1)];
    }
  xc[idx] = siluf_(s);
}

// ---------------- cross-scan: xc(B,DI,L) -> xs(B,K,DI,L) f32 and xsT(B,K,L,DI) f16 ----------------
__global__ void cross_scan_kernel(const float* __restrict__ xc,
                                  float* __restrict__ xs,
                                  _Float16* __restrict__ xsT) {
  int idx = blockIdx.x * blockDim.x + threadIdx.x;
  if (idx >= B_ * DI_ * L_) return;
  int l = idx % L_;
  int d = (idx / L_) % DI_;
  int b = idx / (L_ * DI_);
  float v = xc[idx];
  _Float16 vh = (_Float16)v;
  int i = l >> 5, j = l & 31;
  int lt = j * GS_ + i;
  int pos0 = l, pos1 = lt, pos2 = L_ - 1 - l, pos3 = L_ - 1 - lt;
  // channel-major (for selective scan)
  xs[(((long)b * K_ + 0) * DI_ + d) * L_ + pos0] = v;
  xs[(((long)b * K_ + 1) * DI_ + d) * L_ + pos1] = v;
  xs[(((long)b * K_ + 2) * DI_ + d) * L_ + pos2] = v;
  xs[(((long)b * K_ + 3) * DI_ + d) * L_ + pos3] = v;
  // token-major f16 (K-contiguous rows for x_proj WMMA GEMM)
  xsT[(((long)b * K_ + 0) * L_ + pos0) * DI_ + d] = vh;
  xsT[(((long)b * K_ + 1) * L_ + pos1) * DI_ + d] = vh;
  xsT[(((long)b * K_ + 2) * L_ + pos2) * DI_ + d] = vh;
  xsT[(((long)b * K_ + 3) * L_ + pos3) * DI_ + d] = vh;
}

// ---------------- dt projection + softplus ----------------
__global__ void dtproj_kernel(const float* __restrict__ dbl,
                              const float* __restrict__ dt_w,
                              const float* __restrict__ dt_b,
                              float* __restrict__ dt) {
  long idx = (long)blockIdx.x * blockDim.x + threadIdx.x;
  const long total = (long)B_ * K_ * DI_ * L_;
  if (idx >= total) return;
  int  l  = (int)(idx % L_);
  long r1 = idx / L_;
  int  d  = (int)(r1 % DI_);
  long bk = r1 / DI_;
  int  k  = (int)(bk % K_);
  const float* row  = dbl + bk * (long)(W44_ * L_) + l;
  const float* wrow = dt_w + ((long)(k * DI_ + d)) * R_;
  float s = dt_b[k * DI_ + d];
#pragma unroll
  for (int r = 0; r < R_; ++r) s += row[(long)r * L_] * wrow[r];
  dt[idx] = (s > 20.f) ? s : log1pf(__expf(s));   // softplus
}

// ---------------- selective scan (one lane per (b,k,d) channel) ----------------
__global__ void selscan_kernel(const float* __restrict__ xs,
                               const float* __restrict__ dt,
                               const float* __restrict__ dbl,
                               const float* __restrict__ A_log,
                               const float* __restrict__ Dskip,
                               float* __restrict__ ys) {
  int idx = blockIdx.x * blockDim.x + threadIdx.x;
  if (idx >= B_ * K_ * DI_) return;
  int d  = idx % DI_;
  int bk = idx / DI_;
  int k  = bk % K_;

  const float* up  = xs + (long)idx * L_;
  const float* dtp = dt + (long)idx * L_;
  const float* Bs  = dbl + ((long)bk * W44_ + R_) * L_;
  const float* Cs  = dbl + ((long)bk * W44_ + R_ + DS_) * L_;
  float*       yp  = ys + (long)idx * L_;

  float Arow[DS_];
#pragma unroll
  for (int n = 0; n < DS_; ++n)
    Arow[n] = -__expf(A_log[((long)(k * DI_ + d)) * DS_ + n]);
  float Dsk = Dskip[k * DI_ + d];

  float h[DS_];
#pragma unroll
  for (int n = 0; n < DS_; ++n) h[n] = 0.f;

  for (int l = 0; l < L_; ++l) {
    float dtt = dtp[l];
    float ut  = up[l];
    float du  = dtt * ut;
    float y = 0.f;
#pragma unroll
    for (int n = 0; n < DS_; ++n) {
      float dA = __expf(dtt * Arow[n]);
      h[n] = h[n] * dA + du * Bs[(long)n * L_ + l];
      y += h[n] * Cs[(long)n * L_ + l];
    }
    yp[l] = y + Dsk * ut;
  }
}

// ---------------- cross-merge: ys(B,K,DI,L) -> y(B,L,DI) ----------------
__global__ void cross_merge_kernel(const float* __restrict__ ys, float* __restrict__ y) {
  int idx = blockIdx.x * blockDim.x + threadIdx.x;
  if (idx >= B_ * L_ * DI_) return;
  int d = idx % DI_;
  int l = (idx / DI_) % L_;
  int b = idx / (DI_ * L_);
  int i = l >> 5, j = l & 31;
  int lt = j * GS_ + i;
  float v = ys[(((long)b * K_ + 0) * DI_ + d) * L_ + l];
  v +=      ys[(((long)b * K_ + 1) * DI_ + d) * L_ + lt];
  v +=      ys[(((long)b * K_ + 2) * DI_ + d) * L_ + (L_ - 1 - l)];
  v +=      ys[(((long)b * K_ + 3) * DI_ + d) * L_ + (L_ - 1 - lt)];
  y[idx] = v;
}

// ---------------- gating: yg(f16) = yln * silu(z) ----------------
__global__ void gate_kernel(const float* __restrict__ yln, const float* __restrict__ xz,
                            _Float16* __restrict__ yg) {
  int idx = blockIdx.x * blockDim.x + threadIdx.x;
  if (idx >= B_ * L_ * DI_) return;
  int d  = idx % DI_;
  long t = idx / DI_;
  float z = xz[t * (2 * DI_) + DI_ + d];
  yg[idx] = (_Float16)(yln[idx] * siluf_(z));
}

// ---------------- elementwise add ----------------
__global__ void add_kernel(const float* __restrict__ a, const float* __restrict__ b,
                           float* __restrict__ c, int n) {
  int idx = blockIdx.x * blockDim.x + threadIdx.x;
  if (idx < n) c[idx] = a[idx] + b[idx];
}

// ---------------- window gather (f16): roll(-shift) + partition ----------------
__global__ void win_gather_kernel(const _Float16* __restrict__ x, _Float16* __restrict__ xw,
                                  int shift) {
  int idx = blockIdx.x * blockDim.x + threadIdx.x;
  if (idx >= B_ * L_ * D_) return;
  int d  = idx % D_;
  int rp = idx / D_;
  int p  = rp % 64;
  int w  = (rp / 64) % NW_;
  int b  = rp / (64 * NW_);
  int pi = p >> 3, pj = p & 7;
  int wi = w >> 2, wj = w & 3;
  int gi = (wi * WS_ + pi + shift) & 31;
  int gj = (wj * WS_ + pj + shift) & 31;
  xw[idx] = x[((long)(b * L_ + gi * GS_ + gj)) * D_ + d];
}

// ---------------- window attention core (per b,window,head) ----------------
__device__ __forceinline__ int region3_(int c) { return (c < GS_ - WS_) ? 0 : ((c < GS_ - SHIFT_) ? 1 : 2); }

__global__ void winattn_kernel(const float* __restrict__ qkv,
                               const float* __restrict__ rpb,
                               _Float16* __restrict__ o, int shift) {
  int bwh = blockIdx.x;
  int h  = bwh % NH_;
  int bw = bwh / NH_;
  int w  = bw % NW_;
  int b  = bw / NW_;

  __shared__ float Ks[64][HD_];
  __shared__ float Vs[64][HD_ + 1];

  const int q = threadIdx.x;  // 0..63
  const float* base = qkv + ((long)(b * NW_ + w) * 64) * (3 * D_);

  {
    int p = q;
    for (int hd = 0; hd < HD_; ++hd) {
      Ks[p][hd] = base[(long)p * (3 * D_) + (NH_ + h) * HD_ + hd];
      Vs[p][hd] = base[(long)p * (3 * D_) + (2 * NH_ + h) * HD_ + hd];
    }
  }
  __syncthreads();

  float Q[HD_];
  for (int hd = 0; hd < HD_; ++hd) Q[hd] = base[(long)q * (3 * D_) + h * HD_ + hd];

  const float scale = rsqrtf((float)HD_);
  int qi = q >> 3, qj = q & 7;
  int wi = w >> 2, wj = w & 3;

  float att[64];
  float mx = -1e30f;
  for (int kk = 0; kk < 64; ++kk) {
    float s = 0.f;
    for (int hd = 0; hd < HD_; ++hd) s += Q[hd] * Ks[kk][hd];
    s *= scale;
    int ki = kk >> 3, kj = kk & 7;
    int rel = (qi - ki + WS_ - 1) * (2 * WS_ - 1) + (qj - kj + WS_ - 1);
    s += rpb[rel * NH_ + h];
    if (shift) {
      int rq = region3_(wi * WS_ + qi) * 3 + region3_(wj * WS_ + qj);
      int rk = region3_(wi * WS_ + ki) * 3 + region3_(wj * WS_ + kj);
      if (rq != rk) s += -100.f;
    }
    att[kk] = s;
    mx = fmaxf(mx, s);
  }
  float den = 0.f;
  for (int kk = 0; kk < 64; ++kk) { att[kk] = __expf(att[kk] - mx); den += att[kk]; }
  float inv = 1.f / den;

  _Float16* orow = o + ((long)(b * NW_ + w) * 64 + q) * D_ + h * HD_;
  for (int hd = 0; hd < HD_; ++hd) {
    float s = 0.f;
    for (int kk = 0; kk < 64; ++kk) s += att[kk] * Vs[kk][hd];
    orow[hd] = (_Float16)(s * inv);
  }
}

// ---------------- window scatter (reverse + roll(+shift)) with residual add ----------------
__global__ void win_scatter_add_kernel(const float* __restrict__ pbuf,
                                       const float* __restrict__ res,
                                       float* __restrict__ out, int shift) {
  int idx = blockIdx.x * blockDim.x + threadIdx.x;
  if (idx >= B_ * L_ * D_) return;
  int d = idx % D_;
  int t = (idx / D_) % L_;
  int b = idx / (D_ * L_);
  int gi = t >> 5, gj = t & 31;
  int ri = (gi - shift) & 31, rj = (gj - shift) & 31;
  int wi = ri >> 3, pi = ri & 7, wj = rj >> 3, pj = rj & 7;
  long row = (long)(b * NW_ + wi * 4 + wj) * 64 + (pi * 8 + pj);
  out[idx] = res[idx] + pbuf[row * D_ + d];
}

// ---------------- fusion ----------------
__global__ void fusion_kernel(const float* __restrict__ low, const float* __restrict__ high,
                              const float* __restrict__ fw, float* __restrict__ out) {
  int idx = blockIdx.x * blockDim.x + threadIdx.x;
  if (idx >= B_ * L_ * D_) return;
  float w = sigmoidf_(fw[0]);
  out[idx] = w * low[idx] + (1.f - w) * high[idx];
}

// ==================== host-side launch ====================
static inline void launch_gemm(hipStream_t s, const _Float16* A, const _Float16* W,
                               const float* bias, float* C, _Float16* Ch,
                               int M, int N, int K, int lda,
                               int scM, int scN,
                               int batch, long aStr, long cStr,
                               int wMod, long wStr, int act) {
  dim3 grid((unsigned)(M / 32), (unsigned)((N + 31) / 32), (unsigned)batch);
  gemm_wmma_kernel<<<grid, 32, 0, s>>>(A, W, bias, C, Ch, M, N, K, lda, scM, scN,
                                       aStr, cStr, wMod, wStr, act);
}

#define EW(n) dim3(((n) + 255) / 256), dim3(256)

extern "C" void kernel_launch(void* const* d_in, const int* in_sizes, int n_in,
                              void* d_out, int out_size, void* d_ws, size_t ws_size,
                              hipStream_t stream) {
  const float* const* in = (const float* const*)d_in;
  (void)in_sizes; (void)n_in; (void)out_size; (void)ws_size;

  // input indices (setup_inputs insertion order)
  const float* x_in = in[0];
  const int PE_LOW = 1, PE_HIGH = 6;   // w,b,ng,nb,alpha
  const int MAMBA0 = 11;               // 19 tensors per block
  const int SWIN0  = 49;               // 15 tensors per block
  const int FUSW   = 79;

  // workspace arena (256B-aligned buffers)
  char* base = (char*)d_ws;
  size_t off = 0;
  auto allocf = [&](size_t n) -> float* {
    off = (off + 255) & ~(size_t)255;
    float* p = (float*)(base + off); off += n * sizeof(float); return p;
  };
  auto alloch = [&](size_t n) -> _Float16* {
    off = (off + 255) & ~(size_t)255;
    _Float16* p = (_Float16*)(base + off); off += n * sizeof(_Float16); return p;
  };

  // f32 buffers
  float* pos   = allocf((size_t)L_ * D_);
  float* low   = allocf((size_t)B_ * L_ * D_);
  float* high  = allocf((size_t)B_ * L_ * D_);
  float* xtmp  = allocf((size_t)B_ * L_ * D_);
  float* xz    = allocf((size_t)B_ * L_ * 2 * DI_);
  float* xc    = allocf((size_t)B_ * DI_ * L_);
  float* xs    = allocf((size_t)B_ * K_ * DI_ * L_);
  float* dbl   = allocf((size_t)B_ * K_ * W44_ * L_);
  float* dtb   = allocf((size_t)B_ * K_ * DI_ * L_);
  float* ysb   = allocf((size_t)B_ * K_ * DI_ * L_);
  float* ym    = allocf((size_t)B_ * L_ * DI_);
  float* yln   = allocf((size_t)B_ * L_ * DI_);
  float* tmp1  = allocf((size_t)B_ * L_ * D_);
  float* qkvb  = allocf((size_t)B_ * L_ * 3 * D_);
  float* xpre  = allocf((size_t)B_ * L_ * D_);
  float* t1    = allocf((size_t)B_ * L_ * D_);
  // f16 activation buffers (GEMM A operands)
  _Float16* xnh   = alloch((size_t)B_ * L_ * D_);
  _Float16* xsT   = alloch((size_t)B_ * K_ * L_ * DI_);
  _Float16* ygate = alloch((size_t)B_ * L_ * DI_);
  _Float16* hbuf  = alloch((size_t)B_ * L_ * 4 * D_);
  _Float16* xw    = alloch((size_t)B_ * L_ * D_);
  _Float16* obuf  = alloch((size_t)B_ * L_ * D_);
  // f16 weight staging (reused per block)
  _Float16* wh_in    = alloch((size_t)2 * DI_ * D_);      // 768x192
  _Float16* wh_xproj = alloch((size_t)K_ * W44_ * DI_);   // 4x44x384
  _Float16* wh_out   = alloch((size_t)D_ * DI_);          // 192x384
  _Float16* wh_fc1   = alloch((size_t)4 * D_ * D_);       // 768x192
  _Float16* wh_fc2   = alloch((size_t)D_ * 4 * D_);       // 192x768
  _Float16* wh_qkv   = alloch((size_t)3 * D_ * D_);       // 576x192
  _Float16* wh_proj  = alloch((size_t)D_ * D_);           // 192x192

  const int nBLD  = B_ * L_ * D_;
  const int nBLDI = B_ * L_ * DI_;
  const int nBKDL = B_ * K_ * DI_ * L_;
  const int ROWS = B_ * L_;            // 2048
  const int LNBLK = 256;               // 8 waves per block
  const int lnGrid = (ROWS + (LNBLK / 32) - 1) / (LNBLK / 32);

  // positional embedding
  pos_kernel<<<EW(L_ * D_), 0, stream>>>(pos);

  // patch embeds: conv -> LN(+alpha*pos)
  patch_conv_kernel<<<EW(nBLD), 0, stream>>>(x_in, in[PE_LOW + 0], in[PE_LOW + 1], xtmp);
  ln_kernel<<<lnGrid, LNBLK, 0, stream>>>(xtmp, in[PE_LOW + 2], in[PE_LOW + 3], low, nullptr,
                                          ROWS, D_, pos, in[PE_LOW + 4], L_);
  patch_conv_kernel<<<EW(nBLD), 0, stream>>>(x_in, in[PE_HIGH + 0], in[PE_HIGH + 1], xtmp);
  ln_kernel<<<lnGrid, LNBLK, 0, stream>>>(xtmp, in[PE_HIGH + 2], in[PE_HIGH + 3], high, nullptr,
                                          ROWS, D_, pos, in[PE_HIGH + 4], L_);

  // ---------------- Mamba (VSS) branch ----------------
  for (int mb = 0; mb < 2; ++mb) {
    const int P0 = MAMBA0 + mb * 19;
    const float *ln1g = in[P0+0], *ln1b = in[P0+1], *in_w = in[P0+2];
    const float *conv_w = in[P0+3], *conv_b = in[P0+4], *xproj_w = in[P0+5];
    const float *dt_w = in[P0+6], *dt_b = in[P0+7], *A_log = in[P0+8], *Dsk = in[P0+9];
    const float *ong = in[P0+10], *onb = in[P0+11], *out_w = in[P0+12];
    const float *ln2g = in[P0+13], *ln2b = in[P0+14];
    const float *fc1w = in[P0+15], *fc1b = in[P0+16], *fc2w = in[P0+17], *fc2b = in[P0+18];

    // stage weights to f16
    f2h_kernel<<<EW(2 * DI_ * D_), 0, stream>>>(in_w, wh_in, 2 * DI_ * D_);
    f2h_kernel<<<EW(K_ * W44_ * DI_), 0, stream>>>(xproj_w, wh_xproj, K_ * W44_ * DI_);
    f2h_kernel<<<EW(D_ * DI_), 0, stream>>>(out_w, wh_out, D_ * DI_);
    f2h_kernel<<<EW(4 * D_ * D_), 0, stream>>>(fc1w, wh_fc1, 4 * D_ * D_);
    f2h_kernel<<<EW(D_ * 4 * D_), 0, stream>>>(fc2w, wh_fc2, D_ * 4 * D_);

    // ss2d(ln(x))
    ln_kernel<<<lnGrid, LNBLK, 0, stream>>>(low, ln1g, ln1b, nullptr, xnh,
                                            ROWS, D_, nullptr, nullptr, 1);
    launch_gemm(stream, xnh, wh_in, nullptr, xz, nullptr, ROWS, 2 * DI_, D_, D_,
                2 * DI_, 1, 1, 0, 0, 1, 0, 0);
    dwconv_silu_kernel<<<EW(B_ * DI_ * L_), 0, stream>>>(xz, conv_w, conv_b, xc);
    cross_scan_kernel<<<EW(B_ * DI_ * L_), 0, stream>>>(xc, xs, xsT);
    // x_proj: 8 batched GEMMs (b,k): dbl[bk,c,l] = sum_d xsT[bk,l,d] * xproj_w[k,c,d]
    launch_gemm(stream, xsT, wh_xproj, nullptr, dbl, nullptr, L_, W44_, DI_, DI_,
                1, L_, B_ * K_, (long)L_ * DI_, (long)W44_ * L_,
                K_, (long)W44_ * DI_, 0);
    dtproj_kernel<<<EW(nBKDL), 0, stream>>>(dbl, dt_w, dt_b, dtb);
    selscan_kernel<<<EW(B_ * K_ * DI_), 0, stream>>>(xs, dtb, dbl, A_log, Dsk, ysb);
    cross_merge_kernel<<<EW(nBLDI), 0, stream>>>(ysb, ym);
    ln_kernel<<<lnGrid, LNBLK, 0, stream>>>(ym, ong, onb, yln, nullptr,
                                            ROWS, DI_, nullptr, nullptr, 1);
    gate_kernel<<<EW(nBLDI), 0, stream>>>(yln, xz, ygate);
    launch_gemm(stream, ygate, wh_out, nullptr, tmp1, nullptr, ROWS, D_, DI_, DI_,
                D_, 1, 1, 0, 0, 1, 0, 0);
    add_kernel<<<EW(nBLD), 0, stream>>>(low, tmp1, low, nBLD);

    // MLP
    ln_kernel<<<lnGrid, LNBLK, 0, stream>>>(low, ln2g, ln2b, nullptr, xnh,
                                            ROWS, D_, nullptr, nullptr, 1);
    launch_gemm(stream, xnh, wh_fc1, fc1b, nullptr, hbuf, ROWS, 4 * D_, D_, D_,
                4 * D_, 1, 1, 0, 0, 1, 0, 1 /*gelu*/);
    launch_gemm(stream, hbuf, wh_fc2, fc2b, tmp1, nullptr, ROWS, D_, 4 * D_, 4 * D_,
                D_, 1, 1, 0, 0, 1, 0, 0);
    add_kernel<<<EW(nBLD), 0, stream>>>(low, tmp1, low, nBLD);
  }

  // ---------------- Swin branch ----------------
  for (int sb = 0; sb < 2; ++sb) {
    const int P0 = SWIN0 + sb * 15;
    const int shift = (sb % 2 == 0) ? 0 : SHIFT_;
    const float *preg = in[P0+0], *preb = in[P0+1], *n1g = in[P0+2], *n1b = in[P0+3];
    const float *qkvw = in[P0+4], *qkvbias = in[P0+5], *rpb = in[P0+6];
    const float *projw = in[P0+7], *projb = in[P0+8];
    const float *n2g = in[P0+9], *n2b = in[P0+10];
    const float *fc1w = in[P0+11], *fc1b = in[P0+12], *fc2w = in[P0+13], *fc2b = in[P0+14];

    // stage weights to f16
    f2h_kernel<<<EW(3 * D_ * D_), 0, stream>>>(qkvw, wh_qkv, 3 * D_ * D_);
    f2h_kernel<<<EW(D_ * D_), 0, stream>>>(projw, wh_proj, D_ * D_);
    f2h_kernel<<<EW(4 * D_ * D_), 0, stream>>>(fc1w, wh_fc1, 4 * D_ * D_);
    f2h_kernel<<<EW(D_ * 4 * D_), 0, stream>>>(fc2w, wh_fc2, D_ * 4 * D_);

    ln_kernel<<<lnGrid, LNBLK, 0, stream>>>(high, preg, preb, xpre, nullptr,
                                            ROWS, D_, nullptr, nullptr, 1);
    // attention: t1 = xpre + attn(ln(xpre))
    ln_kernel<<<lnGrid, LNBLK, 0, stream>>>(xpre, n1g, n1b, nullptr, xnh,
                                            ROWS, D_, nullptr, nullptr, 1);
    win_gather_kernel<<<EW(nBLD), 0, stream>>>(xnh, xw, shift);
    launch_gemm(stream, xw, wh_qkv, qkvbias, qkvb, nullptr, ROWS, 3 * D_, D_, D_,
                3 * D_, 1, 1, 0, 0, 1, 0, 0);
    winattn_kernel<<<B_ * NW_ * NH_, 64, 0, stream>>>(qkvb, rpb, obuf, shift);
    launch_gemm(stream, obuf, wh_proj, projb, tmp1, nullptr, ROWS, D_, D_, D_,
                D_, 1, 1, 0, 0, 1, 0, 0);
    win_scatter_add_kernel<<<EW(nBLD), 0, stream>>>(tmp1, xpre, t1, shift);
    // MLP: t1 += mlp(ln(t1))
    ln_kernel<<<lnGrid, LNBLK, 0, stream>>>(t1, n2g, n2b, nullptr, xnh,
                                            ROWS, D_, nullptr, nullptr, 1);
    launch_gemm(stream, xnh, wh_fc1, fc1b, nullptr, hbuf, ROWS, 4 * D_, D_, D_,
                4 * D_, 1, 1, 0, 0, 1, 0, 1 /*gelu*/);
    launch_gemm(stream, hbuf, wh_fc2, fc2b, tmp1, nullptr, ROWS, D_, 4 * D_, 4 * D_,
                D_, 1, 1, 0, 0, 1, 0, 0);
    add_kernel<<<EW(nBLD), 0, stream>>>(t1, tmp1, t1, nBLD);
    // outer residual: high = swin_block(ln(high)) + high
    add_kernel<<<EW(nBLD), 0, stream>>>(t1, high, high, nBLD);
  }

  fusion_kernel<<<EW(nBLD), 0, stream>>>(low, high, in[FUSW], (float*)d_out);
}
